// Mamba2Block_13804024889693
// MI455X (gfx1250) — compile-verified, matrix operand order checked by reading, not run on hardware
//
#include <hip/hip_runtime.h>
#include <hip/hip_bf16.h>
#include <math.h>

// ---- problem constants (from reference) ----
#define D_MODEL_ 1024
#define D_INNER_ 2048
#define D_STATE_ 64
#define D_CONV_  4
#define T_LEN_   1024
#define NB_      2
#define BT_      (NB_ * T_LEN_)      // 2048 rows

typedef __bf16 bf16_t;
typedef bf16_t v16bf __attribute__((ext_vector_type(16)));
typedef float  v8f   __attribute__((ext_vector_type(8)));
typedef int    gvec4i __attribute__((vector_size(16)));   // matches builtin param type

__device__ __forceinline__ bf16_t f2bf(float f) {
  unsigned u = __float_as_uint(f);
  unsigned r = (u + 0x7fffu + ((u >> 16) & 1u)) >> 16;   // round-to-nearest-even
  return __builtin_bit_cast(bf16_t, (unsigned short)r);
}
__device__ __forceinline__ float sigmoidf_(float x) { return 1.0f / (1.0f + __expf(-x)); }
__device__ __forceinline__ float siluf_(float x)    { return x * sigmoidf_(x); }
__device__ __forceinline__ float softplusf_(float x){ return (x > 20.0f) ? x : log1pf(__expf(x)); }

// ---- gfx1250 async global->LDS copy (ASYNCcnt path), with safe fallback ----
#if __has_builtin(__builtin_amdgcn_global_load_async_to_lds_b128)
#define HAVE_ASYNC_G2L 1
#else
#define HAVE_ASYNC_G2L 0
#endif

__device__ __forceinline__ void g2l_b128(void* lds_dst, const void* gsrc) {
#if HAVE_ASYNC_G2L
  __builtin_amdgcn_global_load_async_to_lds_b128(
      (__attribute__((address_space(1))) gvec4i*)(void*)gsrc,
      (__attribute__((address_space(3))) gvec4i*)lds_dst, 0, 0);
#else
  *(uint4*)lds_dst = *(const uint4*)gsrc;
#endif
}
__device__ __forceinline__ void wait_g2l() {
#if HAVE_ASYNC_G2L && __has_builtin(__builtin_amdgcn_s_wait_asynccnt)
  __builtin_amdgcn_s_wait_asynccnt(0);
#elif HAVE_ASYNC_G2L
  asm volatile("s_wait_asynccnt 0x0" ::: "memory");
#endif
}

// ---------------- tiled transpose + f32->bf16: Wt[n][k] = bf16(W[k][n]) ----------------
__global__ __launch_bounds__(256) void transpose_cvt_kernel(const float* __restrict__ W,
                                                            bf16_t* __restrict__ Wt,
                                                            int K, int N) {
  __shared__ float tile[32][33];
  int tx = threadIdx.x & 31;
  int ty = threadIdx.x >> 5;                 // 0..7
  int n0 = blockIdx.x * 32, k0 = blockIdx.y * 32;
  #pragma unroll
  for (int j = 0; j < 32; j += 8)
    tile[ty + j][tx] = W[(size_t)(k0 + ty + j) * N + n0 + tx];
  __syncthreads();
  #pragma unroll
  for (int j = 0; j < 32; j += 8)
    Wt[(size_t)(n0 + ty + j) * K + k0 + tx] = f2bf(tile[tx][ty + j]);
}

// ---------------- LayerNorm (one block per row), emits bf16 ----------------
__global__ __launch_bounds__(256) void ln_bf16_kernel(const float* __restrict__ x,
                                                      const float* __restrict__ g,
                                                      const float* __restrict__ b,
                                                      bf16_t* __restrict__ xn) {
  __shared__ float red[256];
  int row = blockIdx.x, tid = threadIdx.x;
  const float* xr = x + (size_t)row * D_MODEL_;
  float v[4];
  float s = 0.f;
  #pragma unroll
  for (int i = 0; i < 4; i++) { v[i] = xr[tid + i * 256]; s += v[i]; }
  red[tid] = s; __syncthreads();
  for (int st = 128; st > 0; st >>= 1) { if (tid < st) red[tid] += red[tid + st]; __syncthreads(); }
  float mu = red[0] * (1.0f / D_MODEL_);
  __syncthreads();
  float sq = 0.f;
  #pragma unroll
  for (int i = 0; i < 4; i++) { float d = v[i] - mu; sq += d * d; }
  red[tid] = sq; __syncthreads();
  for (int st = 128; st > 0; st >>= 1) { if (tid < st) red[tid] += red[tid + st]; __syncthreads(); }
  float rstd = rsqrtf(red[0] * (1.0f / D_MODEL_) + 1e-5f);
  bf16_t* o = xn + (size_t)row * D_MODEL_;
  #pragma unroll
  for (int i = 0; i < 4; i++) {
    int c = tid + i * 256;
    o[c] = f2bf((v[i] - mu) * rstd * g[c] + b[c]);
  }
}

// ---------------- WMMA bf16 GEMM: C(f32, MxN) = A(bf16, MxK) * Bt(bf16, NxK)^T ----------------
// A row-major MxK; Bt is the PRE-TRANSPOSED weight (N rows of K) so both tiles stage as
// contiguous b128 chunks. 256 threads = 8 waves (4 M x 2 N); block tile 128x64, K-step 32;
// each wave: 32x32 tile = 2x2 v_wmma_f32_16x16x32_bf16. LDS double-buffered; tiles staged
// with global_load_async_to_lds_b128 (ASYNCcnt) overlapped with WMMA compute.
// Optional resid (MxN, f32) added in the epilogue.
#define BM_ 128
#define BN_ 64
#define BK_ 32

__global__ __launch_bounds__(256) void gemm_wmma_bf16(const bf16_t* __restrict__ A,
                                                      const bf16_t* __restrict__ Bt,
                                                      float* __restrict__ C,
                                                      const float* __restrict__ resid,
                                                      int M, int N, int K) {
  __shared__ __align__(32) bf16_t sA[2][BM_ * BK_];  // [m][k], pitch 32
  __shared__ __align__(32) bf16_t sB[2][BN_ * BK_];  // [n][k], pitch 32

  int tid  = threadIdx.x;
  int lane = tid & 31, wave = tid >> 5;
  int wm = wave & 3, wn = wave >> 2;                 // 4x2 wave grid
  int bm = blockIdx.y * BM_, bn = blockIdx.x * BN_;
  int half = lane >> 4, l15 = lane & 15;

  // staging coordinates (all 128-bit chunks)
  int am0 = tid >> 2,         akc0 = tid & 3;        // A chunks 0..255
  int am1 = (tid + 256) >> 2, akc1 = tid & 3;        // A chunks 256..511
  int bn0 = tid >> 2,         bkc0 = tid & 3;        // B chunks 0..255

  auto stage = [&](int buf, int k0) {
    g2l_b128(&sA[buf][am0 * BK_ + akc0 * 8], &A[(size_t)(bm + am0) * K + k0 + akc0 * 8]);
    g2l_b128(&sA[buf][am1 * BK_ + akc1 * 8], &A[(size_t)(bm + am1) * K + k0 + akc1 * 8]);
    g2l_b128(&sB[buf][bn0 * BK_ + bkc0 * 8], &Bt[(size_t)(bn + bn0) * K + k0 + bkc0 * 8]);
  };

  v8f acc[2][2] = {};

  stage(0, 0);
  wait_g2l();
  __syncthreads();

  int nk = K / BK_;
  for (int kt = 0; kt < nk; kt++) {
    int cur = kt & 1;
    if (kt + 1 < nk) stage(cur ^ 1, (kt + 1) * BK_);   // async copies overlap compute

    // Per-lane K packing (16-bit A 16x32): lanes 0-15 hold K{0..7,16..23},
    // lanes 16-31 hold K{8..15,24..31}. Same pattern for B fragments (col-indexed).
    union { v16bf v; uint4 q[2]; } fa[2], fb[2];
    #pragma unroll
    for (int i = 0; i < 2; i++) {
      int row = wm * 32 + i * 16 + l15;
      fa[i].q[0] = *(const uint4*)&sA[cur][row * BK_ + half * 8];
      fa[i].q[1] = *(const uint4*)&sA[cur][row * BK_ + 16 + half * 8];
    }
    #pragma unroll
    for (int j = 0; j < 2; j++) {
      int col = wn * 32 + j * 16 + l15;
      fb[j].q[0] = *(const uint4*)&sB[cur][col * BK_ + half * 8];
      fb[j].q[1] = *(const uint4*)&sB[cur][col * BK_ + 16 + half * 8];
    }
    #pragma unroll
    for (int i = 0; i < 2; i++)
      #pragma unroll
      for (int j = 0; j < 2; j++)
        acc[i][j] = __builtin_amdgcn_wmma_f32_16x16x32_bf16(
            false, fa[i].v, false, fb[j].v, (short)0, acc[i][j], false, false);

    if (kt + 1 < nk) {
      wait_g2l();
      __syncthreads();
    }
  }

  // C/D layout: VGPR v -> (lanes 0-15: M=v, N=lane) (lanes 16-31: M=8+v, N=lane-16)
  #pragma unroll
  for (int i = 0; i < 2; i++)
    #pragma unroll
    for (int j = 0; j < 2; j++) {
      int trow = bm + wm * 32 + i * 16 + ((lane < 16) ? 0 : 8);
      int tcol = bn + wn * 32 + j * 16 + l15;
      size_t base = (size_t)trow * N + tcol;
      if (resid) {
        #pragma unroll
        for (int v = 0; v < 8; v++) C[base + (size_t)v * N] = acc[i][j][v] + resid[base + (size_t)v * N];
      } else {
        #pragma unroll
        for (int v = 0; v < 8; v++) C[base + (size_t)v * N] = acc[i][j][v];
      }
    }
}

// ---------------- depthwise causal conv (4 taps) + bias + SiLU ----------------
__global__ __launch_bounds__(256) void conv_silu_kernel(const float* __restrict__ xz,
                                                        const float* __restrict__ cw,
                                                        const float* __restrict__ cb,
                                                        float* __restrict__ xcf,
                                                        bf16_t* __restrict__ xcb) {
  int idx = blockIdx.x * 256 + threadIdx.x;     // over BT*D_INNER
  int c = idx & (D_INNER_ - 1);
  int r = idx >> 11;                            // / D_INNER_
  int t = r & (T_LEN_ - 1);
  float acc = cb[c];
  #pragma unroll
  for (int k = 0; k < D_CONV_; k++) {
    int tt = t - (D_CONV_ - 1) + k;
    if (tt >= 0)
      acc += xz[(size_t)(r - (D_CONV_ - 1) + k) * (2 * D_INNER_) + c] * cw[c * D_CONV_ + k];
  }
  float s = siluf_(acc);
  xcf[idx] = s;
  xcb[idx] = f2bf(s);
}

// ---------------- softplus(dt_pre + b_dt) row mean -> dt_mean ----------------
__global__ __launch_bounds__(256) void dtmean_kernel(const float* __restrict__ dtpre,
                                                     const float* __restrict__ bdt,
                                                     float* __restrict__ dtmean) {
  __shared__ float red[256];
  int row = blockIdx.x, tid = threadIdx.x;
  const float* dr = dtpre + (size_t)row * D_INNER_;
  float s = 0.f;
  for (int i = tid; i < D_INNER_; i += 256) s += softplusf_(dr[i] + bdt[i]);
  red[tid] = s; __syncthreads();
  for (int st = 128; st > 0; st >>= 1) { if (tid < st) red[tid] += red[tid + st]; __syncthreads(); }
  if (tid == 0) dtmean[row] = red[0] * (1.0f / D_INNER_);
}

// ---------------- selective scan: one channel per wave, 2 states per lane ----------------
__global__ __launch_bounds__(256) void scan_kernel(const float* __restrict__ xcf,
                                                   const float* __restrict__ bc,
                                                   const float* __restrict__ dtmean,
                                                   const float* __restrict__ A_log,
                                                   float* __restrict__ y) {
  int wave = threadIdx.x >> 5, lane = threadIdx.x & 31;
  int b = blockIdx.x / (D_INNER_ / 8);
  int c = (blockIdx.x % (D_INNER_ / 8)) * 8 + wave;
  int s0 = lane * 2, s1 = lane * 2 + 1;
  float a0 = -__expf(A_log[c * D_STATE_ + s0]);
  float a1 = -__expf(A_log[c * D_STATE_ + s1]);
  float h0 = 0.f, h1 = 0.f;
  for (int t = 0; t < T_LEN_; t++) {
    int row = b * T_LEN_ + t;
    float dtm = dtmean[row];
    float xdt = xcf[(size_t)row * D_INNER_ + c] * dtm;
    const float* bcr = bc + (size_t)row * (2 * D_STATE_);
    float Bv0 = bcr[s0], Bv1 = bcr[s1];
    float Cv0 = bcr[D_STATE_ + s0], Cv1 = bcr[D_STATE_ + s1];
    h0 = __expf(dtm * a0) * h0 + xdt * Bv0;
    h1 = __expf(dtm * a1) * h1 + xdt * Bv1;
    float part = h0 * Cv0 + h1 * Cv1;
    #pragma unroll
    for (int off = 16; off > 0; off >>= 1) part += __shfl_down(part, off, 32);
    if (lane == 0) y[(size_t)row * D_INNER_ + c] = part;
  }
}

// ---------------- y = (y + D*xc) * silu(z), emits bf16 for final GEMM ----------------
__global__ __launch_bounds__(256) void ycomb_kernel(const float* __restrict__ y,
                                                    const float* __restrict__ xcf,
                                                    const float* __restrict__ Dp,
                                                    const float* __restrict__ xz,
                                                    bf16_t* __restrict__ yb) {
  int idx = blockIdx.x * 256 + threadIdx.x;
  int c = idx & (D_INNER_ - 1);
  int r = idx >> 11;
  float z = xz[(size_t)r * (2 * D_INNER_) + D_INNER_ + c];
  float v = (y[idx] + Dp[c] * xcf[idx]) * siluf_(z);
  yb[idx] = f2bf(v);
}

// ===================================================================================
extern "C" void kernel_launch(void* const* d_in, const int* in_sizes, int n_in,
                              void* d_out, int out_size, void* d_ws, size_t ws_size,
                              hipStream_t stream) {
  const float* x      = (const float*)d_in[0];
  const float* W_in   = (const float*)d_in[1];
  const float* conv_w = (const float*)d_in[2];
  const float* conv_b = (const float*)d_in[3];
  const float* W_x    = (const float*)d_in[4];
  const float* W_dt   = (const float*)d_in[5];
  const float* b_dt   = (const float*)d_in[6];
  const float* A_log  = (const float*)d_in[7];
  const float* D_par  = (const float*)d_in[8];
  const float* W_out  = (const float*)d_in[9];
  const float* ln_g   = (const float*)d_in[10];
  const float* ln_b   = (const float*)d_in[11];
  float* out = (float*)d_out;

  char* ws = (char*)d_ws;
  size_t off = 0;
  auto alloc = [&](size_t bytes) -> char* {
    char* p = ws + off;
    off += (bytes + 255) & ~(size_t)255;
    return p;
  };

  // pre-transposed bf16 weights (N x K layouts)
  bf16_t* WinT   = (bf16_t*)alloc((size_t)D_MODEL_ * 2 * D_INNER_ * 2);
  bf16_t* WdtT   = (bf16_t*)alloc((size_t)D_INNER_ * D_INNER_ * 2);
  bf16_t* WxT    = (bf16_t*)alloc((size_t)D_INNER_ * 2 * D_STATE_ * 2);
  bf16_t* WoutT  = (bf16_t*)alloc((size_t)D_INNER_ * D_MODEL_ * 2);
  bf16_t* xn     = (bf16_t*)alloc((size_t)BT_ * D_MODEL_ * 2);
  float*  xz     = (float*) alloc((size_t)BT_ * 2 * D_INNER_ * 4);
  float*  xcf    = (float*) alloc((size_t)BT_ * D_INNER_ * 4);
  bf16_t* xcb    = (bf16_t*)alloc((size_t)BT_ * D_INNER_ * 2);
  float*  dtpre  = (float*) alloc((size_t)BT_ * D_INNER_ * 4);
  float*  dtmean = (float*) alloc((size_t)BT_ * 4);
  float*  bc     = (float*) alloc((size_t)BT_ * 2 * D_STATE_ * 4);
  float*  yscan  = (float*) alloc((size_t)BT_ * D_INNER_ * 4);
  bf16_t* ycomb  = (bf16_t*)alloc((size_t)BT_ * D_INNER_ * 2);
  (void)ws_size; (void)n_in; (void)in_sizes; (void)out_size;

  // weight transpose + bf16 conversion (one-off, tiled through LDS)
  transpose_cvt_kernel<<<dim3((2 * D_INNER_) / 32, D_MODEL_ / 32), 256, 0, stream>>>(
      W_in, WinT, D_MODEL_, 2 * D_INNER_);
  transpose_cvt_kernel<<<dim3(D_INNER_ / 32, D_INNER_ / 32), 256, 0, stream>>>(
      W_dt, WdtT, D_INNER_, D_INNER_);
  transpose_cvt_kernel<<<dim3((2 * D_STATE_) / 32, D_INNER_ / 32), 256, 0, stream>>>(
      W_x, WxT, D_INNER_, 2 * D_STATE_);
  transpose_cvt_kernel<<<dim3(D_MODEL_ / 32, D_INNER_ / 32), 256, 0, stream>>>(
      W_out, WoutT, D_INNER_, D_MODEL_);

  // LayerNorm -> xn (bf16)
  ln_bf16_kernel<<<BT_, 256, 0, stream>>>(x, ln_g, ln_b, xn);

  // GEMM1: xz = xn @ W_in   (2048 x 1024 x 4096)
  gemm_wmma_bf16<<<dim3((2 * D_INNER_) / BN_, BT_ / BM_), 256, 0, stream>>>(
      xn, WinT, xz, nullptr, BT_, 2 * D_INNER_, D_MODEL_);

  // depthwise conv + SiLU -> xcf (f32), xcb (bf16)
  conv_silu_kernel<<<(BT_ * D_INNER_) / 256, 256, 0, stream>>>(xz, conv_w, conv_b, xcf, xcb);

  // GEMM2: dt_pre = xc @ W_dt   (2048 x 2048 x 2048)
  gemm_wmma_bf16<<<dim3(D_INNER_ / BN_, BT_ / BM_), 256, 0, stream>>>(
      xcb, WdtT, dtpre, nullptr, BT_, D_INNER_, D_INNER_);

  // dt_mean = mean(softplus(dt_pre + b_dt))
  dtmean_kernel<<<BT_, 256, 0, stream>>>(dtpre, b_dt, dtmean);

  // GEMM3: bc = xc @ W_x   (2048 x 2048 x 128)
  gemm_wmma_bf16<<<dim3((2 * D_STATE_) / BN_, BT_ / BM_), 256, 0, stream>>>(
      xcb, WxT, bc, nullptr, BT_, 2 * D_STATE_, D_INNER_);

  // selective scan -> yscan
  scan_kernel<<<NB_ * (D_INNER_ / 8), 256, 0, stream>>>(xcf, bc, dtmean, A_log, yscan);

  // gating -> ycomb (bf16)
  ycomb_kernel<<<(BT_ * D_INNER_) / 256, 256, 0, stream>>>(yscan, xcf, D_par, xz, ycomb);

  // GEMM4 (+ fused residual): out = ycomb @ W_out + x   (2048 x 2048 x 1024)
  gemm_wmma_bf16<<<dim3(D_MODEL_ / BN_, BT_ / BM_), 256, 0, stream>>>(
      ycomb, WoutT, out, x, BT_, D_MODEL_, D_INNER_);
}
